// ModelForecast_75900662055146
// MI455X (gfx1250) — compile-verified, matrix operand order checked by reading, not run on hardware
//
#include <hip/hip_runtime.h>
#include <hip/hip_bf16.h>
#include <math.h>

typedef _Float16 f16;
typedef __attribute__((ext_vector_type(16))) _Float16 v16h;
typedef __attribute__((ext_vector_type(8)))  _Float16 v8h;
typedef __attribute__((ext_vector_type(8)))  float    v8f;

#define TB   32
#define TN   64
#define LSEQ 50
#define NSEQ (TB*TN)          // 2048 sequences
#define TT   (NSEQ*LSEQ)      // 102400 tokens
#define DMOD 128
#define DINN 256
#define DST  16
#define DRNK 8
#define NLAY 3

// ---------------------------------------------------------------------------
// Weight conversion: f32 -> f16
__global__ void cvt_f16_kernel(const float* __restrict__ src, f16* __restrict__ dst, int n) {
    int i = blockIdx.x * blockDim.x + threadIdx.x;
    if (i < n) dst[i] = (f16)src[i];
}

// x_proj_W: [NL][40][256] -> padded f16 [NL][48][256] (zero rows 40..47)
__global__ void pad_xproj_kernel(const float* __restrict__ src, f16* __restrict__ dst) {
    int i = blockIdx.x * blockDim.x + threadIdx.x;
    const int total = NLAY * 48 * 256;
    if (i >= total) return;
    int k = i & 255;
    int r = (i >> 8) % 48;
    int l = i / (48 * 256);
    dst[i] = (r < 40) ? (f16)src[(l * 40 + r) * 256 + k] : (f16)0.0f;
}

// ---------------------------------------------------------------------------
// Featurize + MLP: feats(10) -> GELU(64) -> h(128)
__global__ void feat_mlp_kernel(const float* __restrict__ pos, const float* __restrict__ vel,
                                const float* __restrict__ acc, const unsigned char* __restrict__ vmask,
                                const float* __restrict__ W1, const float* __restrict__ b1,
                                const float* __restrict__ W2, const float* __restrict__ b2,
                                float* __restrict__ hout) {
    int t = blockIdx.x * blockDim.x + threadIdx.x;
    if (t >= TT) return;
    float f[10];
    f[0] = pos[t * 2];  f[1] = cosf(pos[t * 2 + 1]);  f[2] = sinf(pos[t * 2 + 1]);
    f[3] = vel[t * 2];  f[4] = cosf(vel[t * 2 + 1]);  f[5] = sinf(vel[t * 2 + 1]);
    f[6] = acc[t * 2];  f[7] = cosf(acc[t * 2 + 1]);  f[8] = sinf(acc[t * 2 + 1]);
    f[9] = vmask[t] ? 1.0f : 0.0f;
    float g[64];
    #pragma unroll 4
    for (int j = 0; j < 64; ++j) {
        float a = b1[j];
        #pragma unroll
        for (int r = 0; r < 10; ++r) a += f[r] * W1[r * 64 + j];
        g[j] = 0.5f * a * (1.0f + erff(a * 0.70710678118654752f));   // exact GELU
    }
    float* hp = hout + (long)t * DMOD;
    #pragma unroll 4
    for (int c = 0; c < DMOD; ++c) {
        float a = b2[c];
        #pragma unroll 8
        for (int j = 0; j < 64; ++j) a += g[j] * W2[j * DMOD + c];
        hp[c] = a;
    }
}

// ---------------------------------------------------------------------------
// res = h (+ residual); u16 = rmsnorm(res) * norm_w   (warp per token)
__global__ void resid_rms_kernel(const float* __restrict__ hin, float* __restrict__ resbuf,
                                 const float* __restrict__ normw, f16* __restrict__ u16,
                                 int add_res) {
    int token = blockIdx.x * 8 + (threadIdx.x >> 5);
    int lane  = threadIdx.x & 31;
    if (token >= TT) return;
    const float* hp = hin + (long)token * DMOD;
    float* rp = resbuf + (long)token * DMOD;
    float v[4]; float ss = 0.0f;
    #pragma unroll
    for (int j = 0; j < 4; ++j) {
        int c = lane + 32 * j;
        float x = hp[c] + (add_res ? rp[c] : 0.0f);
        v[j] = x; ss += x * x;
    }
    #pragma unroll
    for (int off = 16; off >= 1; off >>= 1) ss += __shfl_xor(ss, off, 32);
    float sc = rsqrtf(ss * (1.0f / (float)DMOD) + 1e-5f);
    #pragma unroll
    for (int j = 0; j < 4; ++j) {
        int c = lane + 32 * j;
        rp[c] = v[j];
        u16[(long)token * DMOD + c] = (f16)(v[j] * sc * normw[c]);
    }
}

// ---------------------------------------------------------------------------
// WMMA GEMM: C[M,N] = A[M,K] @ W[N,K]^T  (A,W f16; C f32 accumulate)
// One wave computes a 16 x (16*NT) strip: the A fragment is loaded once per
// K-step and amortized over NT WMMAs (vmem:wmma 2.5:1 instead of 4:1).
// Fragment layout per CDNA5 ISA 7.12.2 (wave32).
template<int NT>
__global__ void gemm_wmma_f16_nt(const f16* __restrict__ A, const f16* __restrict__ Bw,
                                 float* __restrict__ Cf, f16* __restrict__ Ch,
                                 int N, int K, int ngroups_n, int total_groups) {
    int wave = (int)((blockIdx.x * blockDim.x + threadIdx.x) >> 5);
    if (wave >= total_groups) return;
    int lane = threadIdx.x & 31;
    int gn = wave % ngroups_n;            // group of NT consecutive n-tiles
    int tm = wave / ngroups_n;
    int mr = (lane & 15);
    int kb = (lane >> 4) * 8;             // lanes 0-15: K+0..7 ; lanes 16-31: K+8..15
    const f16* arow = A + (long)(tm * 16 + mr) * K;
    const f16* brow[NT];
    #pragma unroll
    for (int j = 0; j < NT; ++j)
        brow[j] = Bw + (long)((gn * NT + j) * 16 + mr) * K;
    v8f c[NT];
    #pragma unroll
    for (int j = 0; j < NT; ++j) c[j] = (v8f){};
    for (int kk = 0; kk < K; kk += 32) {
        v8h a0 = *(const v8h*)(arow + kk + kb);
        v8h a1 = *(const v8h*)(arow + kk + 16 + kb);
        v16h av = __builtin_shufflevector(a0, a1, 0,1,2,3,4,5,6,7,8,9,10,11,12,13,14,15);
        #pragma unroll
        for (int j = 0; j < NT; ++j) {
            v8h b0 = *(const v8h*)(brow[j] + kk + kb);
            v8h b1 = *(const v8h*)(brow[j] + kk + 16 + kb);
            v16h bv = __builtin_shufflevector(b0, b1, 0,1,2,3,4,5,6,7,8,9,10,11,12,13,14,15);
            c[j] = __builtin_amdgcn_wmma_f32_16x16x32_f16(false, av, false, bv, (short)0,
                                                          c[j], false, false);
        }
    }
    // C/D layout: VGPR r -> M = r (lanes 0-15) or 8+r (lanes 16-31); N = lane&15
    int rbase = tm * 16 + ((lane >> 4) * 8);
    #pragma unroll
    for (int j = 0; j < NT; ++j) {
        int col = (gn * NT + j) * 16 + (lane & 15);
        if (Ch) {
            #pragma unroll
            for (int r = 0; r < 8; ++r) Ch[(long)(rbase + r) * N + col] = (f16)c[j][r];
        } else {
            #pragma unroll
            for (int r = 0; r < 8; ++r) Cf[(long)(rbase + r) * N + col] = c[j][r];
        }
    }
}

// ---------------------------------------------------------------------------
// Causal depthwise conv (K=4) + SiLU on first 256 cols of xz
__global__ void conv_silu_kernel(const f16* __restrict__ xz, const float* __restrict__ Wc,
                                 const float* __restrict__ bc, f16* __restrict__ xch) {
    long i = (long)blockIdx.x * blockDim.x + threadIdx.x;
    if (i >= (long)TT * DINN) return;
    int d = (int)(i & 255);
    long token = i >> 8;
    int t = (int)(token % LSEQ);
    long base = token * 512 + d;
    float a = bc[d];
    #pragma unroll
    for (int k = 0; k < 4; ++k) {
        int tt = t - 3 + k;
        if (tt >= 0) a += (float)xz[base + (long)(tt - t) * 512] * Wc[d * 4 + k];
    }
    float s = a / (1.0f + __expf(-a));      // silu
    xch[token * DINN + d] = (f16)s;
}

// ---------------------------------------------------------------------------
// Selective scan: block = one sequence, lane d = one channel, 16 states in regs.
// Fuses dt-projection+softplus, state update, C-contraction, D-skip, SiLU(z) gate.
__global__ void scan_kernel(const float* __restrict__ dbl_, const f16* __restrict__ xch,
                            const f16* __restrict__ xz, const float* __restrict__ Wdt,
                            const float* __restrict__ bdt, const float* __restrict__ Alog,
                            const float* __restrict__ Dskip, f16* __restrict__ y16) {
    int seq = blockIdx.x;
    int d   = threadIdx.x;
    __shared__ float sdbl[40];
    float a_s[DST];
    #pragma unroll
    for (int s = 0; s < DST; ++s) a_s[s] = -__expf(Alog[d * DST + s]);
    float wdt[DRNK];
    #pragma unroll
    for (int r = 0; r < DRNK; ++r) wdt[r] = Wdt[d * DRNK + r];
    float bd  = bdt[d];
    float dsk = Dskip[d];
    float hst[DST];
    #pragma unroll
    for (int s = 0; s < DST; ++s) hst[s] = 0.0f;
    for (int t = 0; t < LSEQ; ++t) {
        long tok = (long)seq * LSEQ + t;
        __syncthreads();
        if (d < 40) sdbl[d] = dbl_[tok * 48 + d];
        __syncthreads();
        float dtp = bd;
        #pragma unroll
        for (int r = 0; r < DRNK; ++r) dtp += sdbl[r] * wdt[r];
        float dtv = (dtp > 20.0f) ? dtp : log1pf(__expf(dtp));   // softplus
        float xt = (float)xch[tok * DINN + d];
        float coef = dtv * xt;
        float y = 0.0f;
        #pragma unroll
        for (int s = 0; s < DST; ++s) {
            float dA = __expf(dtv * a_s[s]);
            hst[s] = dA * hst[s] + coef * sdbl[8 + s];            // B_t = dbl[8..23]
            y += hst[s] * sdbl[24 + s];                           // C_t = dbl[24..39]
        }
        y += dsk * xt;
        float z = (float)xz[tok * 512 + DINN + d];
        y *= z / (1.0f + __expf(-z));                             // * silu(z)
        y16[tok * DINN + d] = (f16)y;
    }
}

// ---------------------------------------------------------------------------
// Final: rmsnorm(h + residual) at t = L-1, apply key mask
__global__ void final_kernel(const float* __restrict__ hin, const float* __restrict__ resbuf,
                             const float* __restrict__ normfw, const unsigned char* __restrict__ keymask,
                             float* __restrict__ out) {
    int q    = blockIdx.x * 8 + (threadIdx.x >> 5);
    int lane = threadIdx.x & 31;
    if (q >= NSEQ) return;
    long tok = (long)q * LSEQ + (LSEQ - 1);
    float v[4]; float ss = 0.0f;
    #pragma unroll
    for (int j = 0; j < 4; ++j) {
        int c = lane + 32 * j;
        float x = hin[tok * DMOD + c] + resbuf[tok * DMOD + c];
        v[j] = x; ss += x * x;
    }
    #pragma unroll
    for (int off = 16; off >= 1; off >>= 1) ss += __shfl_xor(ss, off, 32);
    float sc = rsqrtf(ss * (1.0f / (float)DMOD) + 1e-5f);
    float m = keymask[q] ? 1.0f : 0.0f;
    #pragma unroll
    for (int j = 0; j < 4; ++j) {
        int c = lane + 32 * j;
        out[(long)q * DMOD + c] = v[j] * sc * normfw[c] * m;
    }
}

// ---------------------------------------------------------------------------
extern "C" void kernel_launch(void* const* d_in, const int* in_sizes, int n_in,
                              void* d_out, int out_size, void* d_ws, size_t ws_size,
                              hipStream_t stream) {
    (void)in_sizes; (void)n_in; (void)out_size; (void)ws_size;
    const float* pos    = (const float*)d_in[0];
    const float* vel    = (const float*)d_in[1];
    const float* acc    = (const float*)d_in[2];
    const unsigned char* vmask   = (const unsigned char*)d_in[3];
    const unsigned char* keymask = (const unsigned char*)d_in[4];
    const float* mlp_W1 = (const float*)d_in[5];
    const float* mlp_b1 = (const float*)d_in[6];
    const float* mlp_W2 = (const float*)d_in[7];
    const float* mlp_b2 = (const float*)d_in[8];
    const float* norm_w = (const float*)d_in[9];
    const float* in_proj_W  = (const float*)d_in[10];
    const float* conv_W     = (const float*)d_in[11];
    const float* conv_b     = (const float*)d_in[12];
    const float* x_proj_W   = (const float*)d_in[13];
    const float* dt_proj_W  = (const float*)d_in[14];
    const float* dt_proj_b  = (const float*)d_in[15];
    const float* A_log      = (const float*)d_in[16];
    const float* D_skip     = (const float*)d_in[17];
    const float* out_proj_W = (const float*)d_in[18];
    const float* norm_f_w   = (const float*)d_in[19];
    float* out = (float*)d_out;

    // Workspace carve-out
    char* ws = (char*)d_ws;
    size_t off = 0;
    auto carve = [&](size_t bytes) { char* p = ws + off; off += (bytes + 255) & ~(size_t)255; return p; };
    float* hbuf   = (float*)carve((size_t)TT * DMOD * 4);
    float* resbuf = (float*)carve((size_t)TT * DMOD * 4);
    f16*   u16    = (f16*)  carve((size_t)TT * DMOD * 2);
    f16*   xz16   = (f16*)  carve((size_t)TT * 512  * 2);
    f16*   xch16  = (f16*)  carve((size_t)TT * DINN * 2);
    float* dbl    = (float*)carve((size_t)TT * 48   * 4);
    f16*   y16    = (f16*)  carve((size_t)TT * DINN * 2);
    f16*   w16in  = (f16*)  carve((size_t)NLAY * 512 * DMOD * 2);
    f16*   w16x   = (f16*)  carve((size_t)NLAY * 48  * DINN * 2);
    f16*   w16out = (f16*)  carve((size_t)NLAY * DMOD * DINN * 2);

    // Weight conversion
    {
        int n = NLAY * 512 * DMOD;
        cvt_f16_kernel<<<(n + 255) / 256, 256, 0, stream>>>(in_proj_W, w16in, n);
        n = NLAY * DMOD * DINN;
        cvt_f16_kernel<<<(n + 255) / 256, 256, 0, stream>>>(out_proj_W, w16out, n);
        n = NLAY * 48 * 256;
        pad_xproj_kernel<<<(n + 255) / 256, 256, 0, stream>>>(x_proj_W, w16x);
    }

    // Featurize + MLP -> hbuf
    feat_mlp_kernel<<<(TT + 255) / 256, 256, 0, stream>>>(pos, vel, acc, vmask,
                                                          mlp_W1, mlp_b1, mlp_W2, mlp_b2, hbuf);

    const int mtiles = TT / 16;   // 6400

    for (int i = 0; i < NLAY; ++i) {
        // res = h (+ residual); u16 = rmsnorm(res) * norm_w[i]
        resid_rms_kernel<<<TT / 8, 256, 0, stream>>>(hbuf, resbuf, norm_w + i * DMOD, u16, i > 0);

        // xz[T,512] = u16 @ W_in^T  (WMMA, f16 out; NT=4 -> 8 n-groups)
        {
            int ngn = (512 / 16) / 4, groups = mtiles * ngn;
            gemm_wmma_f16_nt<4><<<groups / 8, 256, 0, stream>>>(
                u16, w16in + (size_t)i * 512 * DMOD, nullptr, xz16, 512, DMOD, ngn, groups);
        }

        // conv + silu -> xch16[T,256]
        conv_silu_kernel<<<(int)(((long)TT * DINN + 255) / 256), 256, 0, stream>>>(
            xz16, conv_W + (size_t)i * DINN * 4, conv_b + (size_t)i * DINN, xch16);

        // dbl[T,48] = xch16 @ W_x^T (padded)  (WMMA, f32 out; NT=3 -> 1 n-group)
        {
            int ngn = 1, groups = mtiles * ngn;
            gemm_wmma_f16_nt<3><<<groups / 8, 256, 0, stream>>>(
                xch16, w16x + (size_t)i * 48 * DINN, dbl, nullptr, 48, DINN, ngn, groups);
        }

        // selective scan (fused dt/softplus/state/gate) -> y16[T,256]
        scan_kernel<<<NSEQ, DINN, 0, stream>>>(dbl, xch16, xz16,
                                               dt_proj_W + (size_t)i * DINN * DRNK,
                                               dt_proj_b + (size_t)i * DINN,
                                               A_log + (size_t)i * DINN * DST,
                                               D_skip + (size_t)i * DINN, y16);

        // h[T,128] = y16 @ W_out^T  (WMMA, f32 out; NT=4 -> 2 n-groups)
        {
            int ngn = (DMOD / 16) / 4, groups = mtiles * ngn;
            gemm_wmma_f16_nt<4><<<groups / 8, 256, 0, stream>>>(
                y16, w16out + (size_t)i * DMOD * DINN, hbuf, nullptr, DMOD, DINN, ngn, groups);
        }
    }

    // Final rmsnorm at t=L-1 + key mask -> out
    final_kernel<<<NSEQ / 8, 256, 0, stream>>>(hbuf, resbuf, norm_f_w, keymask, out);
}
